// TimeResampler_48498770707076
// MI455X (gfx1250) — matrix operand order repaired and checked by reading, not compiled, not run on hardware
//
#include <hip/hip_runtime.h>
#include <cstdint>
#include <cstddef>

typedef __attribute__((ext_vector_type(16))) __bf16 bf16x16;
typedef __attribute__((ext_vector_type(8)))  float  f32x8;
typedef uint32_t u32x4 __attribute__((ext_vector_type(4)));
typedef uint32_t u32x8 __attribute__((ext_vector_type(8)));

union AB32 { uint4 u[2]; bf16x16 v; };

#define WMMA_BF16(a, b, c) \
  __builtin_amdgcn_wmma_f32_16x16x32_bf16(false, (a), false, (b), (short)0, (c), false, false)

// ---------------------------------------------------------------------------
// TDM: DMA one (tile_rows x 32) bf16 tile (row stride K elems) into LDS with
// 16B per-row padding (row stride in LDS = 80B) for conflict-free ds_load_b128.
// D# layout per CDNA5 ISA ch.8 (group0 128b, group1 256b; 2D tile, groups 2/3 NULL).
// ---------------------------------------------------------------------------
__device__ __forceinline__ void tdm_load_tile32(const __bf16* gsrc, uint32_t lds_off,
                                                uint32_t tile_rows, uint32_t Kelem,
                                                uint32_t Nrows) {
  uint64_t ga = (uint64_t)(uintptr_t)gsrc;
  u32x4 g0;
  g0[0] = 1u;                                                  // count=1, user mode
  g0[1] = lds_off;                                             // LDS byte address
  g0[2] = (uint32_t)ga;                                        // global_addr[31:0]
  g0[3] = ((uint32_t)(ga >> 32) & 0x01FFFFFFu) | (2u << 30);   // addr[56:32] | type=2
  u32x8 g1;
  g1[0] = (1u << 16)      // data_size = 2 bytes
        | (1u << 20)      // pad_enable (LDS dest padding)
        | (3u << 22)      // pad_interval: 16 DWORDs (= one 64B tile row)
        | (3u << 25);     // pad_amount: 4 DWORDs (row stride 80B in LDS)
  g1[1] = (Kelem & 0xFFFFu) << 16;                              // tensor_dim0 lo16
  g1[2] = ((Kelem >> 16) & 0xFFFFu) | ((Nrows & 0xFFFFu) << 16);// dim0 hi | dim1 lo
  g1[3] = ((Nrows >> 16) & 0xFFFFu) | (32u << 16);              // dim1 hi | tile_dim0=32
  g1[4] = tile_rows & 0xFFFFu;                                  // tile_dim1 (tile_dim2=0)
  g1[5] = Kelem;                                                // tensor_dim0_stride lo32
  g1[6] = 0u;
  g1[7] = 0u;
  asm volatile("tensor_load_to_lds %0, %1" :: "s"(g0), "s"(g1) : "memory");
}

// ---------------------------------------------------------------------------
// Timestep embedding: t_emb = [cos, sin]; temb = silu(t_emb@W1+b1)@W2+b2
// ---------------------------------------------------------------------------
__global__ void k_time(const float* __restrict__ ts,
                       const float* __restrict__ w1, const float* __restrict__ b1,
                       const float* __restrict__ w2, const float* __restrict__ b2,
                       float* __restrict__ temb) {
  int b = blockIdx.x;
  __shared__ float te[320];
  __shared__ float h1[1024];
  float t = ts[b];
  for (int i = threadIdx.x; i < 160; i += 256) {
    float f = __expf(-logf(10000.f) * (float)i / 160.f);
    float a = t * f;
    te[i] = __cosf(a);
    te[i + 160] = __sinf(a);
  }
  __syncthreads();
  for (int o = threadIdx.x; o < 1024; o += 256) {
    float s = b1[o];
    for (int k = 0; k < 320; ++k) s += te[k] * w1[k * 1024 + o];
    h1[o] = s / (1.f + __expf(-s));
  }
  __syncthreads();
  for (int o = threadIdx.x; o < 1024; o += 256) {
    float s = b2[o];
    for (int k = 0; k < 1024; ++k) s += h1[k] * w2[k * 1024 + o];
    temb[(size_t)b * 1024 + o] = s;
  }
}

// ---------------------------------------------------------------------------
// adaLN: only shift (cols 0..1023) and scale (cols 2048..3071) are used.
// ada2[l][b][0..1023]=shift, [1024..2047]=scale. grid.x = l*32+b
// ---------------------------------------------------------------------------
__global__ void k_ada(const float* __restrict__ temb,
                      const float* __restrict__ aw, const float* __restrict__ ab,
                      float* __restrict__ ada2) {
  int b = blockIdx.x & 31, l = blockIdx.x >> 5;
  __shared__ float st[1024];
  for (int i = threadIdx.x; i < 1024; i += 256) {
    float v = temb[(size_t)b * 1024 + i];
    st[i] = v / (1.f + __expf(-v));
  }
  __syncthreads();
  const float* W = aw + (size_t)l * 1024 * 4096;
  const float* B = ab + (size_t)l * 4096;
  for (int jj = threadIdx.x; jj < 2048; jj += 256) {
    int col = (jj < 1024) ? jj : (jj + 1024);   // scale lives at 2048 + (jj-1024)
    float s = B[col];
    for (int k = 0; k < 1024; ++k) s += st[k] * W[(size_t)k * 4096 + col];
    ada2[((size_t)l * 32 + b) * 2048 + jj] = s;
  }
}

// ---------------------------------------------------------------------------
// f32 -> bf16 elementwise
// ---------------------------------------------------------------------------
__global__ void k_cvt(const float* __restrict__ in, __bf16* __restrict__ out, size_t n) {
  for (size_t i = (size_t)blockIdx.x * 256 + threadIdx.x; i < n; i += (size_t)gridDim.x * 256)
    out[i] = (__bf16)in[i];
}

// ---------------------------------------------------------------------------
// Transpose + cvt: W (L,K,N) f32 -> Wt (L,N,K) bf16; optional per-K row scale.
// grid = (N/32, K/32, L), block = 256
// ---------------------------------------------------------------------------
__global__ void k_tcvt(const float* __restrict__ W, __bf16* __restrict__ Wt,
                       int K, int N, const float* __restrict__ rowscale) {
  int l = blockIdx.z;
  const float* Wl = W + (size_t)l * K * N;
  __bf16* Wtl = Wt + (size_t)l * K * N;
  const float* g = rowscale ? rowscale + (size_t)l * K : nullptr;
  __shared__ float tile[32][33];
  int tx = threadIdx.x & 31, ty = threadIdx.x >> 5;
  int n0 = blockIdx.x * 32, k0 = blockIdx.y * 32;
  for (int j = 0; j < 32; j += 8) {
    int kk = k0 + ty + j;
    float v = Wl[(size_t)kk * N + n0 + tx];
    if (g) v *= g[kk];
    tile[ty + j][tx] = v;
  }
  __syncthreads();
  for (int j = 0; j < 32; j += 8)
    Wtl[(size_t)(n0 + ty + j) * K + k0 + tx] = (__bf16)tile[tx][ty + j];
}

// ---------------------------------------------------------------------------
// kv bias': kvb[l][j] = ln1_b[l] . wkv[l][:,j]   (j < 2048). grid=(8,8)
// ---------------------------------------------------------------------------
__global__ void k_kvbias(const float* __restrict__ b1, const float* __restrict__ wkv,
                         float* __restrict__ kvb) {
  int l = blockIdx.y;
  int j = blockIdx.x * 256 + threadIdx.x;
  const float* W = wkv + (size_t)l * 1024 * 2048;
  const float* bb = b1 + (size_t)l * 1024;
  float s = 0.f;
  for (int k = 0; k < 1024; ++k) s += bb[k] * W[(size_t)k * 2048 + j];
  kvb[(size_t)l * 2048 + j] = s;
}

// ---------------------------------------------------------------------------
// Row LayerNorm (width 1024) f32 -> bf16. pad16: rows with (row%16)>=8 -> 0.
// null g/b => gamma=1, beta=0. grid = rows, block = 256
// ---------------------------------------------------------------------------
__global__ void k_ln(const float* __restrict__ X, const float* __restrict__ g,
                     const float* __restrict__ b, __bf16* __restrict__ Y, int pad16) {
  int row = blockIdx.x;
  if (pad16 && (row & 15) >= 8) {
    for (int c = threadIdx.x; c < 1024; c += 256) Y[(size_t)row * 1024 + c] = (__bf16)0.f;
    return;
  }
  const float* x = X + (size_t)row * 1024;
  float v[4], s = 0.f, s2 = 0.f;
#pragma unroll
  for (int i = 0; i < 4; ++i) { v[i] = x[threadIdx.x + i * 256]; s += v[i]; s2 += v[i] * v[i]; }
  __shared__ float rs[256], rq[256];
  rs[threadIdx.x] = s; rq[threadIdx.x] = s2;
  __syncthreads();
  for (int o = 128; o; o >>= 1) {
    if ((int)threadIdx.x < o) { rs[threadIdx.x] += rs[threadIdx.x + o]; rq[threadIdx.x] += rq[threadIdx.x + o]; }
    __syncthreads();
  }
  float mean = rs[0] * (1.f / 1024.f);
  float var = rq[0] * (1.f / 1024.f) - mean * mean;
  float inv = rsqrtf(var + 1e-5f);
#pragma unroll
  for (int i = 0; i < 4; ++i) {
    int c = threadIdx.x + i * 256;
    float gg = g ? g[c] : 1.f, bb = b ? b[c] : 0.f;
    Y[(size_t)row * 1024 + c] = (__bf16)((v[i] - mean) * inv * gg + bb);
  }
}

// ---------------------------------------------------------------------------
// Final LN -> d_out (compact to (b, q<8)). grid = 512 rows
// ---------------------------------------------------------------------------
__global__ void k_final_ln(const float* __restrict__ X, const float* __restrict__ g,
                           const float* __restrict__ b, float* __restrict__ out) {
  int row = blockIdx.x;
  int bq = row & 15, bb_ = row >> 4;
  if (bq >= 8) return;
  const float* x = X + (size_t)row * 1024;
  float v[4], s = 0.f, s2 = 0.f;
#pragma unroll
  for (int i = 0; i < 4; ++i) { v[i] = x[threadIdx.x + i * 256]; s += v[i]; s2 += v[i] * v[i]; }
  __shared__ float rs[256], rq[256];
  rs[threadIdx.x] = s; rq[threadIdx.x] = s2;
  __syncthreads();
  for (int o = 128; o; o >>= 1) {
    if ((int)threadIdx.x < o) { rs[threadIdx.x] += rs[threadIdx.x + o]; rq[threadIdx.x] += rq[threadIdx.x + o]; }
    __syncthreads();
  }
  float mean = rs[0] * (1.f / 1024.f);
  float var = rq[0] * (1.f / 1024.f) - mean * mean;
  float inv = rsqrtf(var + 1e-5f);
#pragma unroll
  for (int i = 0; i < 4; ++i) {
    int c = threadIdx.x + i * 256;
    out[((size_t)bb_ * 8 + bq) * 1024 + c] = (v[i] - mean) * inv * g[c] + b[c];
  }
}

// ---------------------------------------------------------------------------
// lat init: broadcast latents into padded (32,16,1024) f32, rows q>=8 zero
// ---------------------------------------------------------------------------
__global__ void k_fill_lat(const float* __restrict__ latents, float* __restrict__ lat) {
  int row = blockIdx.x, q = row & 15;
  for (int c = threadIdx.x; c < 1024; c += 256)
    lat[(size_t)row * 1024 + c] = (q < 8) ? latents[(size_t)q * 1024 + c] : 0.f;
}

// ---------------------------------------------------------------------------
// Generic bf16 WMMA GEMM: C(MxN) = A(MxK) @ Bt(NxK)^T, fused epilogues.
// 8 waves, block tile 64(M) x 128(N); wave tile 16x64 (4 frags).
// A and B k-slabs staged in LDS by the Tensor Data Mover (double buffered,
// 80B padded rows, TENSORcnt sync). Inner loop: 10 grouped ds_load_b128
// (pinned above the WMMAs by a sched barrier) -> 4 WMMAs.
// grid = (N/128, M/64), block = 256
// ---------------------------------------------------------------------------
#define GM_F32  0
#define GM_GELU 3
#define GM_RES  4
#define GM_Q    5
#define GM_KV   6

#define BROW 40                      // LDS row stride in bf16 elems (80B)
#define ASTAGE (64 * BROW)           // A stage: 64 rows
#define BSTAGE (128 * BROW)          // B stage: 128 rows

__global__ void k_gemm(const __bf16* __restrict__ A, const __bf16* __restrict__ Bt,
                       int M, int N, int K, int mode,
                       const float* __restrict__ bias,
                       float* __restrict__ Cf, __bf16* __restrict__ Cb,
                       const float* __restrict__ res,
                       __bf16* __restrict__ Kb, __bf16* __restrict__ Vb,
                       const float* __restrict__ ada) {
  __shared__ __bf16 sA[2][ASTAGE];
  __shared__ __bf16 sB[2][BSTAGE];
  int w = threadIdx.x >> 5, lane = threadIdx.x & 31;
  int half = (lane >> 4) & 1, lc = lane & 15;
  int wm = w & 3, wn = w >> 2;
  int bm0 = blockIdx.y * 64;
  int bn0 = blockIdx.x * 128;
  uint32_t ldsA = (uint32_t)(size_t)(&sA[0][0]);
  uint32_t ldsB = (uint32_t)(size_t)(&sB[0][0]);
  const int NT = K >> 5;
  const __bf16* atile = A + (size_t)bm0 * K;
  const __bf16* btile = Bt + (size_t)bn0 * K;

  f32x8 acc[4] = {};
  if (w == 0) {
    tdm_load_tile32(atile, ldsA, 64, (uint32_t)K, (uint32_t)M);
    tdm_load_tile32(btile, ldsB, 128, (uint32_t)K, (uint32_t)N);
  }

  for (int it = 0; it < NT; ++it) {
    if (w == 0) __builtin_amdgcn_s_wait_tensorcnt(0);
    __syncthreads();
    // kick off next k-slab DMA; it overlaps this step's LDS reads + WMMAs
    if (w == 0 && it + 1 < NT) {
      uint32_t st = (uint32_t)(((it + 1) & 1));
      tdm_load_tile32(atile + (size_t)(it + 1) * 32, ldsA + st * (ASTAGE * 2), 64, K, M);
      tdm_load_tile32(btile + (size_t)(it + 1) * 32, ldsB + st * (BSTAGE * 2), 128, K, N);
    }
    const __bf16* sa = &sA[it & 1][0];
    const __bf16* sb = &sB[it & 1][0];
    // gather all fragments first; sched barrier keeps loads above the WMMAs
    AB32 a, b0, b1, b2, b3;
    const __bf16* ap = sa + (wm * 16 + lc) * BROW + half * 8;
    a.u[0] = *(const uint4*)(ap);
    a.u[1] = *(const uint4*)(ap + 16);
    const __bf16* bp = sb + (wn * 64 + lc) * BROW + half * 8;
    b0.u[0] = *(const uint4*)(bp);
    b0.u[1] = *(const uint4*)(bp + 16);
    b1.u[0] = *(const uint4*)(bp + 16 * BROW);
    b1.u[1] = *(const uint4*)(bp + 16 * BROW + 16);
    b2.u[0] = *(const uint4*)(bp + 32 * BROW);
    b2.u[1] = *(const uint4*)(bp + 32 * BROW + 16);
    b3.u[0] = *(const uint4*)(bp + 48 * BROW);
    b3.u[1] = *(const uint4*)(bp + 48 * BROW + 16);
    __builtin_amdgcn_sched_barrier(0);
    acc[0] = WMMA_BF16(a.v, b0.v, acc[0]);
    acc[1] = WMMA_BF16(a.v, b1.v, acc[1]);
    acc[2] = WMMA_BF16(a.v, b2.v, acc[2]);
    acc[3] = WMMA_BF16(a.v, b3.v, acc[3]);
  }

  int m0 = bm0 + wm * 16;
  int n0 = bn0 + wn * 64;
#pragma unroll
  for (int i = 0; i < 4; ++i) {
#pragma unroll
    for (int r = 0; r < 8; ++r) {
      int gm = m0 + r + half * 8;
      int gn = n0 + i * 16 + lc;
      float v = acc[i][r];
      if (mode == GM_F32) {
        if (bias) v += bias[gn];
        Cf[(size_t)gm * N + gn] = v;
      } else if (mode == GM_GELU) {
        v = 0.5f * v * (1.f + erff(v * 0.70710678118654752f));
        Cb[(size_t)gm * N + gn] = (__bf16)v;
      } else if (mode == GM_RES) {
        Cf[(size_t)gm * N + gn] = res[(size_t)gm * N + gn] + v;
      } else if (mode == GM_Q) {
        int b = gm >> 4, q = gm & 15;
        int h = gn >> 6, d = gn & 63;
        float shift = ada[(size_t)b * 2048 + gn];
        float scale = ada[(size_t)b * 2048 + 1024 + gn];
        float qv = (q < 8) ? (v * (1.f + scale) + shift) * 0.125f : 0.f;  // DH^-0.5 folded
        Cb[(((size_t)b * 16 + h) * 16 + q) * 64 + d] = (__bf16)qv;
      } else { // GM_KV
        int b = gm >> 10, t = gm & 1023;
        v += bias[gn];
        if (gn < 1024) {
          int h = gn >> 6, d = gn & 63;
          Kb[(((size_t)b * 16 + h) * 1024 + t) * 64 + d] = (__bf16)v;
        } else {
          int j = gn - 1024, h = j >> 6, d = j & 63;
          Vb[(((size_t)b * 16 + h) * 64 + d) * 1024 + t] = (__bf16)v;
        }
      }
    }
  }
}

// ---------------------------------------------------------------------------
// Fused attention per (b,h): S=q@K^T (WMMA), softmax (wave32 shuffles),
// O=P@V (WMMA from LDS). grid = (16 heads, 32 batch), block = 256
// ---------------------------------------------------------------------------
__global__ void k_attn(const __bf16* __restrict__ qbuf, const __bf16* __restrict__ Kbuf,
                       const __bf16* __restrict__ Vt, __bf16* __restrict__ obuf) {
  int h = blockIdx.x, b = blockIdx.y;
  int w = threadIdx.x >> 5, lane = threadIdx.x & 31;
  int half = (lane >> 4) & 1, lc = lane & 15;
  __shared__ float sS[8 * 1024];
  __shared__ __bf16 sP[16 * 1024];

  const __bf16* qp = qbuf + (((size_t)b * 16 + h) * 16) * 64;
  const __bf16* kp = Kbuf + ((size_t)b * 16 + h) * 1024 * 64;
  const __bf16* vp = Vt + ((size_t)b * 16 + h) * 64 * 1024;

  // Phase 1: S[8,1024] = (q*SCALE) @ K^T   (each wave: 128 tokens)
  {
    // q fragments are tf-invariant: hoist both k-steps (k0=0,32) out of the loop
    AB32 q0, q1;
    const __bf16* qp0 = qp + (size_t)lc * 64 + half * 8;
    q0.u[0] = *(const uint4*)(qp0);
    q0.u[1] = *(const uint4*)(qp0 + 16);
    q1.u[0] = *(const uint4*)(qp0 + 32);
    q1.u[1] = *(const uint4*)(qp0 + 48);
    int tw = w * 128;
    for (int tf = 0; tf < 8; ++tf) {
      int n0 = tw + tf * 16;
      AB32 c0, c1;
      const __bf16* brow = kp + (size_t)(n0 + lc) * 64 + half * 8;
      c0.u[0] = *(const uint4*)(brow);
      c0.u[1] = *(const uint4*)(brow + 16);
      c1.u[0] = *(const uint4*)(brow + 32);
      c1.u[1] = *(const uint4*)(brow + 48);
      __builtin_amdgcn_sched_barrier(0);
      f32x8 acc = {};
      acc = WMMA_BF16(q0.v, c0.v, acc);
      acc = WMMA_BF16(q1.v, c1.v, acc);
      if (!half) {
#pragma unroll
        for (int r = 0; r < 8; ++r) sS[r * 1024 + n0 + lc] = acc[r];
      }
    }
  }
  // warm V (phase-3 operand) into cache while softmax runs
  for (int i = threadIdx.x; i < 1024; i += 256)
    __builtin_prefetch(vp + (size_t)i * 64, 0, 0);
  __syncthreads();

  // Phase 2: softmax rows (wave w handles row w), write normalized P (bf16, padded)
  {
    float* row = sS + w * 1024;
    float m = -1e30f;
    for (int t = lane; t < 1024; t += 32) m = fmaxf(m, row[t]);
    for (int off = 16; off; off >>= 1) m = fmaxf(m, __shfl_xor(m, off, 32));
    float s = 0.f;
    for (int t = lane; t < 1024; t += 32) { float e = __expf(row[t] - m); row[t] = e; s += e; }
    for (int off = 16; off; off >>= 1) s += __shfl_xor(s, off, 32);
    float inv = 1.f / s;
    for (int t = lane; t < 1024; t += 32) {
      sP[w * 1024 + t] = (__bf16)(row[t] * inv);
      sP[(w + 8) * 1024 + t] = (__bf16)0.f;
    }
  }
  __syncthreads();

  // Phase 3: O[16,64] = P @ V  (waves 0..3, one 16-wide d-slab each);
  // unroll by 2: both k-steps' fragments loaded before the chained WMMAs
  if (w < 4) {
    int n0 = w * 16;
    f32x8 acc = {};
    for (int k0 = 0; k0 < 1024; k0 += 64) {
      AB32 a0, a1, c0, c1;
      const __bf16* sp0 = sP + (size_t)lc * 1024 + k0 + half * 8;
      a0.u[0] = *(const uint4*)(sp0);
      a0.u[1] = *(const uint4*)(sp0 + 16);
      a1.u[0] = *(const uint4*)(sp0 + 32);
      a1.u[1] = *(const uint4*)(sp0 + 48);
      const __bf16* vr = vp + (size_t)(n0 + lc) * 1024 + k0 + half * 8;
      c0.u[0] = *(const uint4*)(vr);
      c0.u[1] = *(const uint4*)(vr + 16);
      c1.u[0] = *(const uint4*)(vr + 32);
      c1.u[1] = *(const uint4*)(vr + 48);
      __builtin_amdgcn_sched_barrier(0);
      acc = WMMA_BF16(a0.v, c0.v, acc);
      acc = WMMA_BF16(a1.v, c1.v, acc);
    }
#pragma unroll
    for (int r = 0; r < 8; ++r) {
      int q = r + half * 8;
      float v = (q < 8) ? acc[r] : 0.f;
      obuf[((size_t)b * 16 + q) * 1024 + h * 64 + n0 + lc] = (__bf16)v;
    }
  }
}

// ---------------------------------------------------------------------------
// Host orchestration
// ---------------------------------------------------------------------------
static inline char* ws_take(char*& p, size_t bytes) {
  char* r = p;
  p += (bytes + 255) & ~(size_t)255;
  return r;
}

extern "C" void kernel_launch(void* const* d_in, const int* in_sizes, int n_in,
                              void* d_out, int out_size, void* d_ws, size_t ws_size,
                              hipStream_t stream) {
  (void)in_sizes; (void)n_in; (void)out_size; (void)ws_size;
  const float* x      = (const float*)d_in[0];
  const float* tstep  = (const float*)d_in[1];
  const float* lat0   = (const float*)d_in[2];
  const float* te_w1  = (const float*)d_in[3];
  const float* te_b1  = (const float*)d_in[4];
  const float* te_w2  = (const float*)d_in[5];
  const float* te_b2  = (const float*)d_in[6];
  const float* pin_w  = (const float*)d_in[7];
  const float* pin_b  = (const float*)d_in[8];
  const float* ln1_g  = (const float*)d_in[9];
  const float* ln1_b  = (const float*)d_in[10];
  const float* ln2_g  = (const float*)d_in[11];
  const float* ln2_b  = (const float*)d_in[12];
  const float* wq     = (const float*)d_in[13];
  const float* wkv    = (const float*)d_in[14];
  const float* wo     = (const float*)d_in[15];
  const float* ffln_g = (const float*)d_in[16];
  const float* ffln_b = (const float*)d_in[17];
  const float* ffw1   = (const float*)d_in[18];
  const float* ffw2   = (const float*)d_in[19];
  const float* ada_w  = (const float*)d_in[20];
  const float* ada_b  = (const float*)d_in[21];
  const float* pout_w = (const float*)d_in[22];
  const float* pout_b = (const float*)d_in[23];
  const float* nout_g = (const float*)d_in[24];
  const float* nout_b = (const float*)d_in[25];
  float* out = (float*)d_out;

  char* p = (char*)d_ws;
  float*  temb  = (float*) ws_take(p, (size_t)32 * 1024 * 4);
  float*  ada2  = (float*) ws_take(p, (size_t)8 * 32 * 2048 * 4);
  __bf16* xbf   = (__bf16*)ws_take(p, (size_t)32 * 1024 * 768 * 2);
  __bf16* pint  = (__bf16*)ws_take(p, (size_t)1024 * 768 * 2);
  float*  xp    = (float*) ws_take(p, (size_t)32768 * 1024 * 4);
  __bf16* xhat  = (__bf16*)ws_take(p, (size_t)32768 * 1024 * 2);
  __bf16* wkvt  = (__bf16*)ws_take(p, (size_t)8 * 2048 * 1024 * 2);
  float*  kvb   = (float*) ws_take(p, (size_t)8 * 2048 * 4);
  __bf16* wqt   = (__bf16*)ws_take(p, (size_t)8 * 1024 * 1024 * 2);
  __bf16* wot   = (__bf16*)ws_take(p, (size_t)8 * 1024 * 1024 * 2);
  __bf16* f1t   = (__bf16*)ws_take(p, (size_t)8 * 4096 * 1024 * 2);
  __bf16* f2t   = (__bf16*)ws_take(p, (size_t)8 * 1024 * 4096 * 2);
  __bf16* poutt = (__bf16*)ws_take(p, (size_t)1024 * 1024 * 2);
  __bf16* Kbuf  = (__bf16*)ws_take(p, (size_t)32 * 16 * 1024 * 64 * 2);
  __bf16* Vtb   = (__bf16*)ws_take(p, (size_t)32 * 16 * 64 * 1024 * 2);
  __bf16* qbuf  = (__bf16*)ws_take(p, (size_t)32 * 16 * 16 * 64 * 2);
  __bf16* latn  = (__bf16*)ws_take(p, (size_t)512 * 1024 * 2);
  float*  lat   = (float*) ws_take(p, (size_t)512 * 1024 * 4);
  __bf16* obuf  = (__bf16*)ws_take(p, (size_t)512 * 1024 * 2);
  __bf16* hff   = (__bf16*)ws_take(p, (size_t)512 * 1024 * 2);
  __bf16* h1    = (__bf16*)ws_take(p, (size_t)512 * 4096 * 2);
  __bf16* latbf = (__bf16*)ws_take(p, (size_t)512 * 1024 * 2);
  float*  potmp = (float*) ws_take(p, (size_t)512 * 1024 * 4);

  // --- prologue ---
  k_time<<<32, 256, 0, stream>>>(tstep, te_w1, te_b1, te_w2, te_b2, temb);
  k_ada<<<256, 256, 0, stream>>>(temb, ada_w, ada_b, ada2);

  k_cvt<<<4096, 256, 0, stream>>>(x, xbf, (size_t)32 * 1024 * 768);
  k_tcvt<<<dim3(32, 24, 1), 256, 0, stream>>>(pin_w, pint, 768, 1024, nullptr);
  k_tcvt<<<dim3(64, 32, 8), 256, 0, stream>>>(wkv, wkvt, 1024, 2048, ln1_g);
  k_kvbias<<<dim3(8, 8), 256, 0, stream>>>(ln1_b, wkv, kvb);
  k_tcvt<<<dim3(32, 32, 8), 256, 0, stream>>>(wq, wqt, 1024, 1024, nullptr);
  k_tcvt<<<dim3(32, 32, 8), 256, 0, stream>>>(wo, wot, 1024, 1024, nullptr);
  k_tcvt<<<dim3(128, 32, 8), 256, 0, stream>>>(ffw1, f1t, 1024, 4096, nullptr);
  k_tcvt<<<dim3(32, 128, 8), 256, 0, stream>>>(ffw2, f2t, 4096, 1024, nullptr);
  k_tcvt<<<dim3(32, 32, 1), 256, 0, stream>>>(pout_w, poutt, 1024, 1024, nullptr);

  // proj_in: xp = x @ pin + b  (M=32768, N=1024, K=768)
  k_gemm<<<dim3(8, 512), 256, 0, stream>>>(xbf, pint, 32768, 1024, 768, GM_F32,
                                           pin_b, xp, nullptr, nullptr, nullptr, nullptr, nullptr);
  // xhat = rownorm(xp)  (affine folded into weights)
  k_ln<<<32768, 256, 0, stream>>>(xp, nullptr, nullptr, xhat, 0);
  k_fill_lat<<<512, 256, 0, stream>>>(lat0, lat);

  // --- layers ---
  for (int l = 0; l < 8; ++l) {
    // kv = xhat @ wkv'(row-scaled) + b@wkv  ->  Kbuf[b,h,t,d], Vt[b,h,d,t]
    k_gemm<<<dim3(16, 512), 256, 0, stream>>>(
        xhat, wkvt + (size_t)l * 2048 * 1024, 32768, 2048, 1024, GM_KV,
        kvb + (size_t)l * 2048, nullptr, nullptr, nullptr, Kbuf, Vtb, nullptr);
    // latn = LN(lat)
    k_ln<<<512, 256, 0, stream>>>(lat, ln2_g + (size_t)l * 1024, ln2_b + (size_t)l * 1024, latn, 1);
    // q = modulate(latn @ wq) * SCALE  ->  qbuf[b,h,16,64]
    k_gemm<<<dim3(8, 8), 256, 0, stream>>>(
        latn, wqt + (size_t)l * 1024 * 1024, 512, 1024, 1024, GM_Q,
        nullptr, nullptr, qbuf, nullptr, nullptr, nullptr, ada2 + (size_t)l * 32 * 2048);
    // attention per (b,h)
    k_attn<<<dim3(16, 32), 256, 0, stream>>>(qbuf, Kbuf, Vtb, obuf);
    // lat += obuf @ wo
    k_gemm<<<dim3(8, 8), 256, 0, stream>>>(
        obuf, wot + (size_t)l * 1024 * 1024, 512, 1024, 1024, GM_RES,
        nullptr, lat, nullptr, lat, nullptr, nullptr, nullptr);
    // hff = LN(lat)
    k_ln<<<512, 256, 0, stream>>>(lat, ffln_g + (size_t)l * 1024, ffln_b + (size_t)l * 1024, hff, 1);
    // h1 = gelu(hff @ ffw1)
    k_gemm<<<dim3(32, 8), 256, 0, stream>>>(
        hff, f1t + (size_t)l * 4096 * 1024, 512, 4096, 1024, GM_GELU,
        nullptr, nullptr, h1, nullptr, nullptr, nullptr, nullptr);
    // lat += h1 @ ffw2
    k_gemm<<<dim3(8, 8), 256, 0, stream>>>(
        h1, f2t + (size_t)l * 1024 * 4096, 512, 1024, 4096, GM_RES,
        nullptr, lat, nullptr, lat, nullptr, nullptr, nullptr);
  }

  // --- epilogue: out = LN(lat @ pout + b) ---
  k_cvt<<<2048, 256, 0, stream>>>(lat, latbf, (size_t)512 * 1024);
  k_gemm<<<dim3(8, 8), 256, 0, stream>>>(latbf, poutt, 512, 1024, 1024, GM_F32,
                                         pout_b, potmp, nullptr, nullptr, nullptr, nullptr, nullptr);
  k_final_ln<<<512, 256, 0, stream>>>(potmp, nout_g, nout_b, out);
}
